// MambaDecoderLayer_53979148976322
// MI455X (gfx1250) — matrix operand order repaired and checked by reading, not compiled, not run on hardware
//
#include <hip/hip_runtime.h>
#include <hip/hip_bf16.h>
#include <math.h>
#include <stdint.h>

typedef __bf16 bf16;
typedef __attribute__((ext_vector_type(16))) __bf16 v16bf;
typedef __attribute__((ext_vector_type(8)))  __bf16 v8bf;
typedef __attribute__((ext_vector_type(8)))  float  v8f;
typedef int v4i __attribute__((vector_size(16)));

#define HD 2048      // H
#define DD 4096      // D
#define NN 16        // state dim
#define RR 128       // dt rank
#define BB 4
#define LL 2048
#define TOK (BB*LL)        // 8192 tokens
#define E_XP (RR + 2*NN)   // 160

// ---------------- workspace layout (bytes) ----------------
static constexpr size_t OFF_XN   = 0;                                  // 8192x2048 bf16
static constexpr size_t OFF_W1   = OFF_XN   + (size_t)TOK*HD*2;        // 8192x2048 bf16
static constexpr size_t OFF_WOUT = OFF_W1   + (size_t)2*DD*HD*2;       // 2048x4096 bf16
static constexpr size_t OFF_WXP  = OFF_WOUT + (size_t)HD*DD*2;         // 160x4096 bf16
static constexpr size_t OFF_WDT  = OFF_WXP  + (size_t)E_XP*DD*2;       // 4096x128 bf16
static constexpr size_t OFF_XS   = OFF_WDT  + (size_t)DD*RR*2;         // 8192x4096 bf16
static constexpr size_t OFF_ZS   = OFF_XS   + (size_t)TOK*DD*2;        // 8192x4096 bf16
static constexpr size_t OFF_XC   = OFF_ZS   + (size_t)TOK*DD*2;        // 8192x4096 bf16
static constexpr size_t OFF_DBL  = OFF_XC   + (size_t)TOK*DD*2;        // 8192x160 f32
static constexpr size_t OFF_DTIN = OFF_DBL  + (size_t)TOK*E_XP*4;      // 8192x128 bf16
static constexpr size_t OFF_DT   = OFF_DTIN + (size_t)TOK*RR*2;        // 8192x4096 f32
static constexpr size_t OFF_Y    = OFF_DT   + (size_t)TOK*DD*4;        // 8192x4096 bf16

// ---------------- async-to-LDS staging (gfx1250 path w/ fallback) ----------------
#if defined(__gfx1250__) && __has_builtin(__builtin_amdgcn_global_load_async_to_lds_b128) && __has_builtin(__builtin_amdgcn_s_wait_asynccnt)
#define USE_ASYNC_LDS 1
#else
#define USE_ASYNC_LDS 0
#endif

typedef __attribute__((address_space(1))) v4i* gv4i_p;
typedef __attribute__((address_space(3))) v4i* lv4i_p;

__device__ inline void stage16(const bf16* g, bf16* l) {
#if USE_ASYNC_LDS
  __builtin_amdgcn_global_load_async_to_lds_b128(
      (gv4i_p)(uintptr_t)g,
      (lv4i_p)(uint32_t)(uintptr_t)l,
      0, 0);
#else
  *(v8bf*)l = *(const v8bf*)g;
#endif
}

template <int N>
__device__ inline void stage_wait() {
#if USE_ASYNC_LDS
  __builtin_amdgcn_s_wait_asynccnt(N);
#endif
}

// ---------------- fp32 -> bf16 conversion ----------------
__global__ __launch_bounds__(256) void cvt_f32_bf16(const float* __restrict__ in,
                                                    bf16* __restrict__ out, long n) {
  long i = (long)blockIdx.x * 256 + threadIdx.x;
  if (i < n) out[i] = (bf16)in[i];
}

// ---------------- fused residual-add + RMSNorm ----------------
__global__ __launch_bounds__(256) void add_rmsnorm_kernel(
    const float* __restrict__ hs, const float* __restrict__ res,
    const float* __restrict__ w, float* __restrict__ res_out,
    bf16* __restrict__ xn) {
  int row = blockIdx.x;
  const float* ph = hs + (size_t)row * HD;
  const float* pr = res + (size_t)row * HD;
  float* po = res_out + (size_t)row * HD;
  bf16* px = xn + (size_t)row * HD;
  float x[8];
  float ss = 0.f;
#pragma unroll
  for (int j = 0; j < 8; ++j) {
    int c = threadIdx.x + j * 256;
    float v = ph[c] + pr[c];
    po[c] = v;
    x[j] = v;
    ss += v * v;
  }
  __shared__ float red[256];
  red[threadIdx.x] = ss;
  __syncthreads();
  for (int s = 128; s > 0; s >>= 1) {
    if (threadIdx.x < s) red[threadIdx.x] += red[threadIdx.x + s];
    __syncthreads();
  }
  float inv = rsqrtf(red[0] * (1.0f / HD) + 1e-5f);
#pragma unroll
  for (int j = 0; j < 8; ++j) {
    int c = threadIdx.x + j * 256;
    px[c] = (bf16)(x[j] * inv * w[c]);
  }
}

// ---------------- epilogue ----------------
struct EpiArgs {
  float* f0;
  bf16* b0;
  bf16* b1;
  const float* bias;
};

// EPI 0: in_proj  -> xs bf16 (col<D), silu(z) bf16 (col>=D)
// EPI 1: x_proj   -> dbl f32, dt_in bf16 (col<R)
// EPI 2: dt_proj  -> softplus(acc + bias) f32
// EPI 3: out_proj -> f32 into d_out
template <int EPI, bool GUARD>
__device__ inline void epilogue(const v8f acc[2][4], int m0, int n0, int N,
                                const EpiArgs& e, unsigned lane) {
  int rofs = (int)((lane >> 4) << 3);
  int ccol = (int)(lane & 15);
#pragma unroll
  for (int i = 0; i < 2; ++i) {
#pragma unroll
    for (int j = 0; j < 4; ++j) {
      union { v8f v; float f[8]; } u;
      u.v = acc[i][j];
#pragma unroll
      for (int r = 0; r < 8; ++r) {
        int row = m0 + i * 16 + r + rofs;
        int col = n0 + j * 16 + ccol;
        if (GUARD && col >= N) continue;
        float v = u.f[r];
        if (EPI == 0) {
          if (col < DD) {
            e.b0[(size_t)row * DD + col] = (bf16)v;
          } else {
            float s = v / (1.f + __expf(-v));
            e.b1[(size_t)row * DD + (col - DD)] = (bf16)s;
          }
        } else if (EPI == 1) {
          e.f0[(size_t)row * E_XP + col] = v;
          if (col < RR) e.b0[(size_t)row * RR + col] = (bf16)v;
        } else if (EPI == 2) {
          float t = v + e.bias[col];
          float sp = (t > 20.f) ? t : log1pf(__expf(t));
          e.f0[(size_t)row * DD + col] = sp;
        } else {
          e.f0[(size_t)row * HD + col] = v;
        }
      }
    }
  }
}

// ---------------- LDS fragment loads ----------------
#define APAD 40  // padded LDS row length in bf16 elements (80B -> 20-bank stride)

__device__ inline v16bf lds_a_frag(const bf16* base, int row0, unsigned lane) {
  const bf16* p = base + (row0 + (int)(lane & 15)) * APAD + (int)((lane >> 4) << 3);
  union { v16bf v; v8bf h[2]; } u;
  u.h[0] = *(const v8bf*)p;
  u.h[1] = *(const v8bf*)(p + 16);
  return u.v;
}

__device__ inline v16bf lds_b_frag(const bf16* base, int row0, unsigned lane) {
  const bf16* p = base + (row0 + (int)(lane & 15)) * APAD + (int)((lane >> 4) << 3);
  union { v16bf v; v8bf h[2]; } u;
  u.h[0] = *(const v8bf*)p;
  u.h[1] = *(const v8bf*)(p + 16);
  return u.v;
}

// ---------------- LDS-staged, double-buffered WMMA GEMM (N % 128 == 0) ----------
template <int EPI>
__global__ __launch_bounds__(256) void gemm_wmma_lds(
    const bf16* __restrict__ A, const bf16* __restrict__ Bm,
    int N, int K, EpiArgs e) {
  __shared__ bf16 sA[2][128 * APAD];
  __shared__ bf16 sB[2][128 * APAD];
  unsigned tid = threadIdx.x, wave = tid >> 5, lane = tid & 31;
  int mblk = blockIdx.y * 128;
  int nblk = blockIdx.x * 128;

  // staging: 512 chunks of 16B per tile; thread t handles chunks t and t+256
  int c0 = (int)tid, c1 = (int)tid + 256;
  int ar0 = c0 >> 2, ap0 = (c0 & 3) << 3;
  int ar1 = c1 >> 2, ap1 = (c1 & 3) << 3;
  const bf16* gA0 = A + (size_t)(mblk + ar0) * K + ap0;
  const bf16* gA1 = A + (size_t)(mblk + ar1) * K + ap1;
  const bf16* gB0 = Bm + (size_t)(nblk + ar0) * K + ap0;
  const bf16* gB1 = Bm + (size_t)(nblk + ar1) * K + ap1;
  bf16* lA0 = &sA[0][ar0 * APAD + ap0];
  bf16* lA1 = &sA[0][ar1 * APAD + ap1];
  bf16* lB0 = &sB[0][ar0 * APAD + ap0];
  bf16* lB1 = &sB[0][ar1 * APAD + ap1];
  const int bufstep = 128 * APAD;

  v8f zacc = {};
  v8f acc[2][4];
#pragma unroll
  for (int i = 0; i < 2; ++i)
#pragma unroll
    for (int j = 0; j < 4; ++j) acc[i][j] = zacc;

  int wm = (int)(wave & 3) * 32;
  int wn = (int)(wave >> 2) * 64;

  // prologue: fill buffer 0
  stage16(gA0, lA0);
  stage16(gA1, lA1);
  stage16(gB0, lB0);
  stage16(gB1, lB1);

  int nk = K / 32;
  for (int kt = 0; kt < nk; ++kt) {
    int cur = kt & 1;
    if (kt + 1 < nk) {
      int nb = (cur ^ 1) * bufstep;
      int ko = (kt + 1) * 32;
      stage16(gA0 + ko, lA0 + nb);
      stage16(gA1 + ko, lA1 + nb);
      stage16(gB0 + ko, lB0 + nb);
      stage16(gB1 + ko, lB1 + nb);
      stage_wait<4>();   // current buffer's 4 issues retired, next 4 in flight
    } else {
      stage_wait<0>();
    }
    __syncthreads();
    const bf16* cA = &sA[cur][0];
    const bf16* cB = &sB[cur][0];
    v16bf a0 = lds_a_frag(cA, wm, lane);
    v16bf a1 = lds_a_frag(cA, wm + 16, lane);
    v16bf b[4];
#pragma unroll
    for (int j = 0; j < 4; ++j) b[j] = lds_b_frag(cB, wn + j * 16, lane);
#pragma unroll
    for (int j = 0; j < 4; ++j) {
      acc[0][j] = __builtin_amdgcn_wmma_f32_16x16x32_bf16(
          false, a0, false, b[j], (short)0, acc[0][j], false, false);
      acc[1][j] = __builtin_amdgcn_wmma_f32_16x16x32_bf16(
          false, a1, false, b[j], (short)0, acc[1][j], false, false);
    }
    __syncthreads();
  }

  epilogue<EPI, false>(acc, mblk + wm, nblk + wn, N, e, lane);
}

// ---------------- direct-global WMMA GEMM with N guards (x_proj, N=160) --------
__device__ inline v16bf load_a_frag(const bf16* __restrict__ A, int lda,
                                    int row0, int k0, unsigned lane) {
  const bf16* p = A + (size_t)(row0 + (int)(lane & 15)) * lda + k0 + (int)((lane >> 4) << 3);
  union { v16bf v; v8bf h[2]; } u;
  u.h[0] = *(const v8bf*)(p);
  u.h[1] = *(const v8bf*)(p + 16);
  return u.v;
}

__device__ inline v16bf load_b_frag(const bf16* __restrict__ Bm, int ldb, int Nrows,
                                    int n0, int k0, unsigned lane) {
  int r = n0 + (int)(lane & 15);
  union { v16bf v; v8bf h[2]; } u;
  u.v = (v16bf){};
  if (r < Nrows) {
    const bf16* p = Bm + (size_t)r * ldb + k0 + (int)((lane >> 4) << 3);
    u.h[0] = *(const v8bf*)(p);
    u.h[1] = *(const v8bf*)(p + 16);
  }
  return u.v;
}

template <int EPI>
__global__ __launch_bounds__(256) void gemm_wmma_direct(
    const bf16* __restrict__ A, const bf16* __restrict__ Bm,
    int N, int K, EpiArgs e) {
  unsigned tid = threadIdx.x, wave = tid >> 5, lane = tid & 31;
  int m0 = blockIdx.y * 128 + (int)(wave & 3) * 32;
  int n0 = blockIdx.x * 128 + (int)(wave >> 2) * 64;

  v8f zacc = {};
  v8f acc[2][4];
#pragma unroll
  for (int i = 0; i < 2; ++i)
#pragma unroll
    for (int j = 0; j < 4; ++j) acc[i][j] = zacc;

  for (int k = 0; k < K; k += 32) {
    v16bf a0 = load_a_frag(A, K, m0, k, lane);
    v16bf a1 = load_a_frag(A, K, m0 + 16, k, lane);
    v16bf b[4];
#pragma unroll
    for (int j = 0; j < 4; ++j) b[j] = load_b_frag(Bm, K, N, n0 + j * 16, k, lane);
#pragma unroll
    for (int j = 0; j < 4; ++j) {
      acc[0][j] = __builtin_amdgcn_wmma_f32_16x16x32_bf16(
          false, a0, false, b[j], (short)0, acc[0][j], false, false);
      acc[1][j] = __builtin_amdgcn_wmma_f32_16x16x32_bf16(
          false, a1, false, b[j], (short)0, acc[1][j], false, false);
    }
  }
  epilogue<EPI, true>(acc, m0, n0, N, e, lane);
}

// ---------------- depthwise causal conv(K=4) + bias + silu ----------------
__global__ __launch_bounds__(256) void conv_silu_kernel(
    const bf16* __restrict__ xs, const float* __restrict__ w,
    const float* __restrict__ bias, bf16* __restrict__ xc) {
  long i = (long)blockIdx.x * 256 + threadIdx.x;  // over TOK*DD
  int d = (int)(i % DD);
  long bl = i / DD;           // token index b*L + t
  int t = (int)(bl % LL);
  float acc = bias[d];
#pragma unroll
  for (int kk = 0; kk < 4; ++kk) {
    int tt = t - 3 + kk;
    if (tt >= 0) acc += w[d * 4 + kk] * (float)xs[(bl - 3 + kk) * DD + d];
  }
  float s = acc / (1.f + __expf(-acc));
  xc[i] = (bf16)s;
}

// ---------------- selective scan ----------------
#define TCH 64
__global__ __launch_bounds__(128) void scan_kernel(
    const float* __restrict__ dt, const bf16* __restrict__ xc,
    const float* __restrict__ dbl, const float* __restrict__ A_log,
    const float* __restrict__ Dp, const bf16* __restrict__ zs,
    bf16* __restrict__ y) {
  int b = blockIdx.x / (DD / 128);
  int d = (blockIdx.x % (DD / 128)) * 128 + (int)threadIdx.x;

  float an[NN], h[NN];
#pragma unroll
  for (int n = 0; n < NN; ++n) {
    an[n] = -__expf(A_log[(size_t)d * NN + n]);
    h[n] = 0.f;
  }
  float dpar = Dp[d];

  __shared__ float sBC[TCH * 32];
  const size_t base = (size_t)b * LL;
  for (int t0 = 0; t0 < LL; t0 += TCH) {
    __syncthreads();
    for (int i = (int)threadIdx.x; i < TCH * 32; i += 128) {
      int tt = i >> 5, c = i & 31;
      sBC[i] = dbl[(base + t0 + tt) * E_XP + RR + c];
    }
    __syncthreads();
    for (int tt = 0; tt < TCH; ++tt) {
      size_t tok = base + t0 + tt;
      float dtv = dt[tok * DD + d];
      float xv = (float)xc[tok * DD + d];
      float dx = dtv * xv;
      const float* Bv = &sBC[tt * 32];
      const float* Cv = Bv + NN;
      float ysum = 0.f;
#pragma unroll
      for (int n = 0; n < NN; ++n) {
        float dA = __expf(dtv * an[n]);
        h[n] = dA * h[n] + dx * Bv[n];
        ysum += h[n] * Cv[n];
      }
      float yv = (ysum + xv * dpar) * (float)zs[tok * DD + d];
      y[tok * DD + d] = (bf16)yv;
    }
  }
}

extern "C" void kernel_launch(void* const* d_in, const int* in_sizes, int n_in,
                              void* d_out, int out_size, void* d_ws, size_t ws_size,
                              hipStream_t stream) {
  const float* hs     = (const float*)d_in[0];
  const float* resid  = (const float*)d_in[1];
  const float* nw     = (const float*)d_in[2];
  const float* w_in   = (const float*)d_in[3];   // (2D, H)
  const float* w_conv = (const float*)d_in[4];   // (D, 1, 4)
  const float* b_conv = (const float*)d_in[5];
  const float* w_xp   = (const float*)d_in[6];   // (160, D)
  const float* w_dt   = (const float*)d_in[7];   // (D, R)
  const float* b_dt   = (const float*)d_in[8];
  const float* a_log  = (const float*)d_in[9];
  const float* d_par  = (const float*)d_in[10];
  const float* w_out  = (const float*)d_in[11];  // (H, D)

  float* out = (float*)d_out;                    // (B,L,H)
  float* res_out = out + (size_t)TOK * HD;       // (B,L,H)

  char* ws = (char*)d_ws;
  bf16*  XN   = (bf16*)(ws + OFF_XN);
  bf16*  W1   = (bf16*)(ws + OFF_W1);
  bf16*  WOUT = (bf16*)(ws + OFF_WOUT);
  bf16*  WXP  = (bf16*)(ws + OFF_WXP);
  bf16*  WDT  = (bf16*)(ws + OFF_WDT);
  bf16*  XS   = (bf16*)(ws + OFF_XS);
  bf16*  ZS   = (bf16*)(ws + OFF_ZS);
  bf16*  XC   = (bf16*)(ws + OFF_XC);
  float* DBL  = (float*)(ws + OFF_DBL);
  bf16*  DTIN = (bf16*)(ws + OFF_DTIN);
  float* DT   = (float*)(ws + OFF_DT);
  bf16*  Y    = (bf16*)(ws + OFF_Y);

  // 1) weight fp32 -> bf16
  {
    long n1 = (long)2 * DD * HD;
    cvt_f32_bf16<<<(unsigned)((n1 + 255) / 256), 256, 0, stream>>>(w_in, W1, n1);
    long n2 = (long)HD * DD;
    cvt_f32_bf16<<<(unsigned)((n2 + 255) / 256), 256, 0, stream>>>(w_out, WOUT, n2);
    long n3 = (long)E_XP * DD;
    cvt_f32_bf16<<<(unsigned)((n3 + 255) / 256), 256, 0, stream>>>(w_xp, WXP, n3);
    long n4 = (long)DD * RR;
    cvt_f32_bf16<<<(unsigned)((n4 + 255) / 256), 256, 0, stream>>>(w_dt, WDT, n4);
  }

  // 2) residual add + rmsnorm
  add_rmsnorm_kernel<<<TOK, 256, 0, stream>>>(hs, resid, nw, res_out, XN);

  // 3) in_proj GEMM (8192 x 8192 x 2048), LDS-staged, fused split/silu epilogue
  {
    EpiArgs e{nullptr, XS, ZS, nullptr};
    gemm_wmma_lds<0><<<dim3(2 * DD / 128, TOK / 128), 256, 0, stream>>>(XN, W1, 2 * DD, HD, e);
  }

  // 4) depthwise causal conv + silu
  conv_silu_kernel<<<(unsigned)(((long)TOK * DD) / 256), 256, 0, stream>>>(XS, w_conv, b_conv, XC);

  // 5) x_proj GEMM (8192 x 160 x 4096), guarded direct path
  {
    EpiArgs e{DBL, DTIN, nullptr, nullptr};
    gemm_wmma_direct<1><<<dim3((E_XP + 127) / 128, TOK / 128), 256, 0, stream>>>(XC, WXP, E_XP, DD, e);
  }

  // 6) dt_proj GEMM (8192 x 4096 x 128) + softplus, LDS-staged
  {
    EpiArgs e{DT, nullptr, nullptr, b_dt};
    gemm_wmma_lds<2><<<dim3(DD / 128, TOK / 128), 256, 0, stream>>>(DTIN, WDT, DD, RR, e);
  }

  // 7) selective scan + gate
  scan_kernel<<<BB * (DD / 128), 128, 0, stream>>>(DT, XC, DBL, a_log, d_par, ZS, Y);

  // 8) out_proj GEMM (8192 x 2048 x 4096) -> d_out, LDS-staged
  {
    EpiArgs e{out, nullptr, nullptr, nullptr};
    gemm_wmma_lds<3><<<dim3(HD / 128, TOK / 128), 256, 0, stream>>>(Y, WOUT, HD, DD, e);
  }
}
